// LSTMLayer_36532991820401
// MI455X (gfx1250) — matrix operand order repaired and checked by reading, not compiled, not run on hardware
//
#include <hip/hip_runtime.h>
#include <math.h>

typedef __bf16 bf16_t;
typedef __attribute__((ext_vector_type(16))) __bf16 v16bf;
typedef __attribute__((ext_vector_type(8)))  __bf16 v8bf;
typedef __attribute__((ext_vector_type(4)))  __bf16 v4bf;
typedef __attribute__((ext_vector_type(8)))  float  v8f;
typedef __attribute__((ext_vector_type(4)))  float  v4f;

static constexpr int Tt = 512;
static constexpr int Bb = 64;
static constexpr int Ii = 1024;
static constexpr int Hh = 1024;
static constexpr int Gg = 4096;   // 4*H

// ---------------- conversion / setup kernels ----------------

__global__ __launch_bounds__(256) void k_cvt_bf16x4(bf16_t* __restrict__ dst,
                                                    const float* __restrict__ src,
                                                    long n4) {
  long gid = (long)blockIdx.x * 256 + threadIdx.x;
  if (gid >= n4) return;
  v4f v = *(const v4f*)(src + gid * 4);
  v4bf o;
  o.x = (bf16_t)v.x; o.y = (bf16_t)v.y; o.z = (bf16_t)v.z; o.w = (bf16_t)v.w;
  *(v4bf*)(dst + gid * 4) = o;
}

__global__ __launch_bounds__(256) void k_bias_sum(float* __restrict__ bias,
                                                  const float* __restrict__ bi,
                                                  const float* __restrict__ bh) {
  int i = blockIdx.x * 256 + threadIdx.x;
  if (i < Gg) bias[i] = bi[i] + bh[i];
}

__global__ __launch_bounds__(256) void k_zero_state(bf16_t* __restrict__ hb,
                                                    float* __restrict__ cst) {
  int i = blockIdx.x * 256 + threadIdx.x;   // B*H = 65536 exact
  hb[i]  = (bf16_t)0.0f;
  cst[i] = 0.0f;
}

// ---------------- WMMA micro-kernel (register double-buffered) ----------------
// acc[MT][NT] += A[MT*16 x K] * W[NT*16 x K]^T ; A,W row-major bf16.
// Fragment layouts used here (wave32; same contiguous k order for A and B):
//  A 16x32: lane m=l&15 holds 16 contiguous K values selected by l>>4 (two 8-bf16 loads).
//  B 32x16: lane n=l&15 holds 16 contiguous K values selected by l>>4.
// k+32 fragments are prefetched before issuing k's WMMAs so s_wait covers
// older loads only (software pipeline depth 1).
template<int MT, int NT>
__device__ inline void wmma_mm(v8f (&acc)[MT][NT],
                               const bf16_t* __restrict__ A, int lda,
                               const bf16_t* __restrict__ W, int ldw,
                               int K, int lane) {
  const int am    = lane & 15;
  const int akoff = (lane >> 4) * 16;
  const int wn    = lane & 15;
  const int wkoff = (lane >> 4) * 16;
  union U { v16bf v; v8bf h[2]; };
  U a[MT]; v16bf b[NT];
#pragma unroll
  for (int i = 0; i < MT; ++i) {
    const bf16_t* ap = A + (size_t)(am + 16 * i) * lda + akoff;
    a[i].h[0] = *(const v8bf*)(ap);
    a[i].h[1] = *(const v8bf*)(ap + 8);
  }
#pragma unroll
  for (int j = 0; j < NT; ++j)
    b[j] = *(const v16bf*)(W + (size_t)(wn + 16 * j) * ldw + wkoff);

  for (int k = 32; k < K; k += 32) {
    U a2[MT]; v16bf b2[NT];
#pragma unroll
    for (int i = 0; i < MT; ++i) {
      const bf16_t* ap = A + (size_t)(am + 16 * i) * lda + k + akoff;
      a2[i].h[0] = *(const v8bf*)(ap);
      a2[i].h[1] = *(const v8bf*)(ap + 8);
    }
#pragma unroll
    for (int j = 0; j < NT; ++j)
      b2[j] = *(const v16bf*)(W + (size_t)(wn + 16 * j) * ldw + k + wkoff);
#pragma unroll
    for (int j = 0; j < NT; ++j)
#pragma unroll
      for (int i = 0; i < MT; ++i)
        acc[i][j] = __builtin_amdgcn_wmma_f32_16x16x32_bf16(
            false, a[i].v, false, b[j], (short)0, acc[i][j], false, false);
#pragma unroll
    for (int i = 0; i < MT; ++i) a[i] = a2[i];
#pragma unroll
    for (int j = 0; j < NT; ++j) b[j] = b2[j];
  }
#pragma unroll
  for (int j = 0; j < NT; ++j)
#pragma unroll
    for (int i = 0; i < MT; ++i)
      acc[i][j] = __builtin_amdgcn_wmma_f32_16x16x32_bf16(
          false, a[i].v, false, b[j], (short)0, acc[i][j], false, false);
}

// ---------------- Phase 1: x_proj = x @ W_ih^T + bias ----------------
// M = T*B = 32768, N = 4096, K = 1024. Wave tile 32x64 (MT=2, NT=4).
__global__ __launch_bounds__(256) void k_xproj_gemm(const bf16_t* __restrict__ xb,
                                                    const bf16_t* __restrict__ wib,
                                                    const float* __restrict__ bias,
                                                    float* __restrict__ xproj) {
  const int wid  = blockIdx.x * 8 + (threadIdx.x >> 5);
  const int lane = threadIdx.x & 31;
  const int nTiles = Gg / 64;                 // 64
  const int m0 = (wid / nTiles) * 32;
  const int n0 = (wid % nTiles) * 64;
  v8f acc[2][4];
#pragma unroll
  for (int i = 0; i < 2; ++i)
#pragma unroll
    for (int j = 0; j < 4; ++j)
#pragma unroll
      for (int r = 0; r < 8; ++r) acc[i][j][r] = 0.0f;

  wmma_mm<2, 4>(acc, xb + (size_t)m0 * Ii, Ii, wib + (size_t)n0 * Ii, Ii, Ii, lane);

  const int lo = lane & 15, hi = lane >> 4;
#pragma unroll
  for (int j = 0; j < 4; ++j) {
    float bv = bias[n0 + 16 * j + lo];
#pragma unroll
    for (int i = 0; i < 2; ++i)
#pragma unroll
      for (int r = 0; r < 8; ++r)
        xproj[(size_t)(m0 + 16 * i + r + 8 * hi) * Gg + n0 + 16 * j + lo] =
            acc[i][j][r] + bv;
  }
}

// ---------------- Fused recurrent step: GEMM + LSTM pointwise ----------------
// One block (8 waves) owns a 32-column span of H for ALL FOUR gates:
//   wave w -> gate g = w>>1, column half = w&1, GEMM row base n0 = g*H + cb + half*16.
// Wave tile 64(M) x 16(N): one B-fragment reused by 4 M-subtiles, so W_hh is
// streamed from L2 exactly once per step. Gate tiles are exchanged via 32 KB
// LDS, then the block applies sigmoid/tanh and updates c/h in-place.
// h is ping-pong buffered (hin -> hout) to avoid a cross-block RAW race.

__device__ inline void step_epilogue(float (*ls)[64][32],
                                     v8f (&acc)[4][1], int g, int half,
                                     int cb, int lane, int tid,
                                     float* __restrict__ cst,
                                     bf16_t* __restrict__ hout,
                                     float* __restrict__ out,
                                     float* __restrict__ out_h,
                                     float* __restrict__ out_c, int t) {
  const int lo = lane & 15, hi = lane >> 4;
#pragma unroll
  for (int i = 0; i < 4; ++i)
#pragma unroll
    for (int r = 0; r < 8; ++r)
      ls[g][16 * i + r + 8 * hi][half * 16 + lo] = acc[i][0][r];
  __syncthreads();
#pragma unroll
  for (int e = 0; e < 8; ++e) {
    const int linear = tid + 256 * e;        // 0..2047
    const int row = linear >> 5;             // batch b: 0..63
    const int col = linear & 31;             // column within block span
    float gi = ls[0][row][col];
    float gf = ls[1][row][col];
    float gg = ls[2][row][col];
    float go = ls[3][row][col];
    float ig = 1.0f / (1.0f + __expf(-gi));
    float fg = 1.0f / (1.0f + __expf(-gf));
    float cg = tanhf(gg);
    float og = 1.0f / (1.0f + __expf(-go));
    const int idx = row * Hh + cb + col;     // (b, j) in [B, H]
    float c = fg * cst[idx] + ig * cg;
    float h = og * tanhf(c);
    cst[idx]  = c;
    hout[idx] = (bf16_t)h;
    out[(size_t)t * (Bb * Hh) + idx] = h;
    if (t == Tt - 1) { out_h[idx] = h; out_c[idx] = c; }
  }
}

__global__ __launch_bounds__(256) void k_step_xproj(const float* __restrict__ xproj,
                                                    const bf16_t* __restrict__ whb,
                                                    const bf16_t* __restrict__ hin,
                                                    bf16_t* __restrict__ hout,
                                                    float* __restrict__ cst,
                                                    float* __restrict__ out,
                                                    float* __restrict__ out_h,
                                                    float* __restrict__ out_c, int t) {
  __shared__ float ls[4][64][32];
  const int tid  = threadIdx.x;
  const int wid  = tid >> 5;
  const int lane = tid & 31;
  const int g = wid >> 1, half = wid & 1;
  const int cb = blockIdx.x * 32;
  const int n0 = g * Hh + cb + half * 16;
  const int lo = lane & 15, hi = lane >> 4;
  v8f acc[4][1];
#pragma unroll
  for (int i = 0; i < 4; ++i)
#pragma unroll
    for (int r = 0; r < 8; ++r)
      acc[i][0][r] = xproj[((size_t)t * Bb + 16 * i + r + 8 * hi) * Gg + n0 + lo];

  wmma_mm<4, 1>(acc, hin, Hh, whb + (size_t)n0 * Hh, Hh, Hh, lane);
  step_epilogue(ls, acc, g, half, cb, lane, tid, cst, hout, out, out_h, out_c, t);
}

__global__ __launch_bounds__(256) void k_step_fused(const bf16_t* __restrict__ xb,
                                                    const bf16_t* __restrict__ wib,
                                                    const bf16_t* __restrict__ whb,
                                                    const bf16_t* __restrict__ hin,
                                                    bf16_t* __restrict__ hout,
                                                    const float* __restrict__ bias,
                                                    float* __restrict__ cst,
                                                    float* __restrict__ out,
                                                    float* __restrict__ out_h,
                                                    float* __restrict__ out_c, int t) {
  __shared__ float ls[4][64][32];
  const int tid  = threadIdx.x;
  const int wid  = tid >> 5;
  const int lane = tid & 31;
  const int g = wid >> 1, half = wid & 1;
  const int cb = blockIdx.x * 32;
  const int n0 = g * Hh + cb + half * 16;
  const int lo = lane & 15;
  float bv = bias[n0 + lo];
  v8f acc[4][1];
#pragma unroll
  for (int i = 0; i < 4; ++i)
#pragma unroll
    for (int r = 0; r < 8; ++r) acc[i][0][r] = bv;

  const bf16_t* xt = xb + (size_t)t * Bb * Ii;
  wmma_mm<4, 1>(acc, xt, Ii, wib + (size_t)n0 * Ii, Ii, Ii, lane);
  wmma_mm<4, 1>(acc, hin, Hh, whb + (size_t)n0 * Hh, Hh, Hh, lane);
  step_epilogue(ls, acc, g, half, cb, lane, tid, cst, hout, out, out_h, out_c, t);
}

// ---------------- host-side orchestration ----------------
extern "C" void kernel_launch(void* const* d_in, const int* in_sizes, int n_in,
                              void* d_out, int out_size, void* d_ws, size_t ws_size,
                              hipStream_t stream) {
  const float* x    = (const float*)d_in[0];
  const float* W_ih = (const float*)d_in[1];
  const float* W_hh = (const float*)d_in[2];
  const float* b_ih = (const float*)d_in[3];
  const float* b_hh = (const float*)d_in[4];

  float* out   = (float*)d_out;
  float* out_h = out + (size_t)Tt * Bb * Hh;
  float* out_c = out_h + (size_t)Bb * Hh;

  char* ws = (char*)d_ws;
  size_t off = 0;
  auto walloc = [&](size_t bytes) -> char* {
    char* p = ws + off;
    off += (bytes + 255) & ~(size_t)255;
    return p;
  };
  bf16_t* xb   = (bf16_t*)walloc((size_t)Tt * Bb * Ii * sizeof(bf16_t));
  bf16_t* wib  = (bf16_t*)walloc((size_t)Gg * Ii * sizeof(bf16_t));
  bf16_t* whb  = (bf16_t*)walloc((size_t)Gg * Hh * sizeof(bf16_t));
  float*  bias = (float*)walloc((size_t)Gg * sizeof(float));
  bf16_t* hb0  = (bf16_t*)walloc((size_t)Bb * Hh * sizeof(bf16_t));
  bf16_t* hb1  = (bf16_t*)walloc((size_t)Bb * Hh * sizeof(bf16_t));
  float*  cst  = (float*)walloc((size_t)Bb * Hh * sizeof(float));
  const size_t base_bytes  = off;
  const size_t xproj_bytes = (size_t)Tt * Bb * Gg * sizeof(float);  // 512 MB
  const bool use_xproj = (ws_size >= base_bytes + xproj_bytes);
  float* xproj = (float*)(ws + base_bytes);

  // Convert inputs to bf16; fold biases; zero initial state (hb0 is first read).
  {
    long n4x = (long)Tt * Bb * Ii / 4;        // 8,388,608
    long n4w = (long)Gg * Ii / 4;             // 1,048,576
    k_cvt_bf16x4<<<(unsigned)((n4x + 255) / 256), 256, 0, stream>>>(xb, x, n4x);
    k_cvt_bf16x4<<<(unsigned)((n4w + 255) / 256), 256, 0, stream>>>(wib, W_ih, n4w);
    k_cvt_bf16x4<<<(unsigned)((n4w + 255) / 256), 256, 0, stream>>>(whb, W_hh, n4w);
    k_bias_sum<<<16, 256, 0, stream>>>(bias, b_ih, b_hh);
    k_zero_state<<<256, 256, 0, stream>>>(hb0, cst);
  }

  if (use_xproj) {
    // One big parallel GEMM: 1024 mTiles * 64 nTiles = 65536 waves.
    k_xproj_gemm<<<8192, 256, 0, stream>>>(xb, wib, bias, xproj);
  }

  for (int t = 0; t < Tt; ++t) {
    bf16_t* hin  = (t & 1) ? hb1 : hb0;
    bf16_t* hout = (t & 1) ? hb0 : hb1;
    if (use_xproj) {
      k_step_xproj<<<32, 256, 0, stream>>>(xproj, whb, hin, hout, cst,
                                           out, out_h, out_c, t);
    } else {
      k_step_fused<<<32, 256, 0, stream>>>(xb, wib, whb, hin, hout, bias, cst,
                                           out, out_h, out_c, t);
    }
  }
}
